// AxialSoftmaxAttention_1864015806925
// MI455X (gfx1250) — compile-verified, matrix operand order checked by reading, not run on hardware
//
#include <hip/hip_runtime.h>
#include <hip/hip_bf16.h>
#include <cstdint>

// ---------------- problem constants ----------------
#define Bv   512
#define Av   32
#define Gv   4
#define Dv   1024
#define Hv   16
#define DKv  64
#define INNERv 1024
#define Pv   129              // A*G + 1
#define MROWS 65536           // B * A * G  (= rows of every big GEMM)

typedef __attribute__((ext_vector_type(16))) __bf16 v16bf;
typedef __attribute__((ext_vector_type(8)))  float  v8f;
typedef __attribute__((ext_vector_type(4)))  unsigned int u32x4;
typedef __attribute__((ext_vector_type(8)))  int i32x8;
typedef __attribute__((ext_vector_type(4)))  int i32x4;

union AFrag { v16bf v; uint32_t u[8]; uint4 q[2]; };

#if __has_builtin(__builtin_amdgcn_tensor_load_to_lds)
#define HAS_TDM 1
#else
#define HAS_TDM 0
#endif

// LDS staging geometry: 64-elem (128B) rows padded to 144B
// (TDM pad: interval 32 DW -> code 4, amount 4 DW -> code 3)
#define LPITCH 72                 // elements per LDS row (144 bytes)
#define ASZ (256 * LPITCH)        // one A buffer (elements)
#define BSZ (128 * LPITCH)        // one B buffer (elements)

// Make TDM-engine LDS writes visible to the compiler: escape the pointer and
// declare memory clobbered, so ds_load fragment reads are not folded to undef.
#define LDS_FENCE(p) asm volatile("" : : "v"((const void*)(p)) : "memory")

// ---------------- fp32 -> bf16 weight convert ----------------
__global__ __launch_bounds__(256) void k_cvt(const float* __restrict__ src,
                                             __bf16* __restrict__ dst, int n) {
    int i = blockIdx.x * 256 + threadIdx.x;
    if (i < n) dst[i] = (__bf16)src[i];
}

// ---------------- x (B,P,D) f32 -> x_ant (B*G, A, D) bf16 ----------------
__global__ __launch_bounds__(256) void k_permute_ant(const float* __restrict__ x,
                                                     __bf16* __restrict__ xa) {
    size_t idx = (size_t)blockIdx.x * 256 + threadIdx.x;   // over 65536*1024
    int d = (int)(idx & 1023);
    size_t row = idx >> 10;          // n*A + a
    int a = (int)(row & 31);
    size_t n = row >> 5;             // b*G + g
    int g = (int)(n & 3);
    size_t b = n >> 2;
    xa[idx] = (__bf16)x[(b * Pv + 1 + (size_t)(a * Gv + g)) * Dv + d];
}

#if HAS_TDM
// Issue one TDM 2D tile load: rows x 64 bf16 elements, row stride K, into LDS
// at byte offset ldsOff with 144B padded row pitch. D# per ISA ch.8.
__device__ __forceinline__ void tdm_stage(const __bf16* gbase, unsigned rowStart,
                                          int k0, int K, unsigned ldsOff,
                                          int tileRows, unsigned tensorRows) {
    unsigned long long ga = (unsigned long long)(uintptr_t)gbase
                          + (((unsigned long long)rowStart * (unsigned)K) + (unsigned)k0) * 2ull;
    u32x4 g0;
    g0.x = 1u;                                           // count=1 (valid descriptor)
    g0.y = ldsOff;                                       // LDS byte address
    g0.z = (unsigned)(ga & 0xffffffffu);                 // global_addr[31:0]
    g0.w = (unsigned)((ga >> 32) & 0x1ffffffu) | (2u << 30);  // global_addr[56:32], type=2
    // group1: data_size=1(2B), pad_enable, pad_interval=4(32DW), pad_amount=3(4DW)
    i32x8 g1 = {
        (int)0x07110000,
        (int)(((unsigned)K & 0xffffu) << 16),                         // tensor_dim0 lo16
        (int)(((unsigned)K >> 16) | ((tensorRows & 0xffffu) << 16)),  // dim0 hi | dim1 lo
        (int)((tensorRows >> 16) | (64u << 16)),                      // dim1 hi | tile_dim0=64
        (int)tileRows,                                                // tile_dim1 | tile_dim2=0
        (int)K,                                                       // tensor_dim0_stride lo32
        0, 0
    };
    i32x4 z4 = {0, 0, 0, 0};
    i32x8 z8 = {0, 0, 0, 0, 0, 0, 0, 0};
    __builtin_amdgcn_tensor_load_to_lds(g0, g1, z4, z4, z8, 0);
}
#endif

// cooperative fallback staging (used if TDM builtin missing)
__device__ __forceinline__ void stage_fallback(const __bf16* __restrict__ src,
                                               __bf16* __restrict__ dst,
                                               int rows, int K, int tid) {
    for (int r = tid; r < rows; r += 256) {
        const uint4* s = (const uint4*)(src + (size_t)r * K);
        uint4* d = (uint4*)(dst + r * LPITCH);
#pragma unroll
        for (int i = 0; i < 8; i++) d[i] = s[i];
    }
}

// ---------------- bf16 WMMA GEMM:  C = A(MxK) * W^T(KxN) + bias ----------------
// W stored row-major (N x K).  mode 0: bf16 out, ldc=N.  mode 1: f32 out into
// d_out with row remap r=(b*128+t) -> (b*129+1+t).
// Block: 256 thr (8 waves, 4x2), block tile 256x128, wave tile 64x64.
// A/B K64-tiles staged to LDS by the Tensor Data Mover, double buffered;
// each staged tile feeds two WMMA k-steps (32 v_wmma per barrier pair).
__global__ __launch_bounds__(256) void k_gemm_bf16(
    const __bf16* __restrict__ A, const __bf16* __restrict__ W,
    const float* __restrict__ bias, void* __restrict__ Cout,
    int M, int N, int K, int mode)
{
    __shared__ __bf16 smem[2 * ASZ + 2 * BSZ];   // A0 A1 B0 B1 ; static LDS base = 0

    const int tid  = threadIdx.x;
    const int lane = tid & 31;
    const int wave = tid >> 5;                  // 0..7
    const int wm = wave >> 1;                   // 0..3
    const int wn = wave & 1;                    // 0..1
    const unsigned bm0 = blockIdx.x * 256;
    const unsigned bn0 = blockIdx.y * 128;
    const int r  = lane & 15;
    const int e0 = (lane >> 4) << 3;            // +8 element chunk offset, upper half-wave
    const bool issuer = (wave == 0);

    v8f acc[4][4];
#pragma unroll
    for (int i = 0; i < 4; i++)
#pragma unroll
        for (int j = 0; j < 4; j++) acc[i][j] = (v8f){0.f,0.f,0.f,0.f,0.f,0.f,0.f,0.f};

    const int NK = K / 64;

#if HAS_TDM
    if (issuer) {
        tdm_stage(A, bm0, 0, K, 0u, 256, (unsigned)M);
        tdm_stage(W, bn0, 0, K, (unsigned)(2 * ASZ) * 2, 128, (unsigned)N);
    }
#else
    stage_fallback(A + (size_t)bm0 * K, smem, 256, K, tid);
    stage_fallback(W + (size_t)bn0 * K, smem + 2 * ASZ, 128, K, tid);
    __syncthreads();
#endif

    for (int it = 0; it < NK; ++it) {
        const int cur = it & 1;
#if HAS_TDM
        if (issuer && (it + 1 < NK)) {
            const int kn = (it + 1) * 64;
            tdm_stage(A, bm0, kn, K, (unsigned)((cur ^ 1) * ASZ) * 2, 256, (unsigned)M);
            tdm_stage(W, bn0, kn, K, (unsigned)(2 * ASZ + (cur ^ 1) * BSZ) * 2, 128, (unsigned)N);
        }
        if (issuer) {
            if (it + 1 < NK) __builtin_amdgcn_s_wait_tensorcnt(2);
            else             __builtin_amdgcn_s_wait_tensorcnt(0);
        }
        __syncthreads();
        LDS_FENCE(smem);
#endif
        const __bf16* as = smem + cur * ASZ;
        const __bf16* bsm = smem + 2 * ASZ + cur * BSZ;

#pragma unroll
        for (int ks = 0; ks < 2; ks++) {
            const int ko = ks * 32;
            AFrag a[4], b[4];
#pragma unroll
            for (int i = 0; i < 4; i++) {
                const int lr = wm * 64 + i * 16 + r;
                a[i].q[0] = *(const uint4*)(as + lr * LPITCH + ko + e0);
                a[i].q[1] = *(const uint4*)(as + lr * LPITCH + ko + 16 + e0);
            }
#pragma unroll
            for (int j = 0; j < 4; j++) {
                const int lr = wn * 64 + j * 16 + r;
                b[j].q[0] = *(const uint4*)(bsm + lr * LPITCH + ko + e0);
                b[j].q[1] = *(const uint4*)(bsm + lr * LPITCH + ko + 16 + e0);
            }
#pragma unroll
            for (int i = 0; i < 4; i++)
#pragma unroll
                for (int j = 0; j < 4; j++)
                    acc[i][j] = __builtin_amdgcn_wmma_f32_16x16x32_bf16(
                        false, a[i].v, false, b[j].v, (short)0, acc[i][j], false, false);
        }

#if HAS_TDM
        __syncthreads();      // all reads of 'cur' done before TDM overwrites it
#else
        __syncthreads();
        if (it + 1 < NK) {
            const int kn = (it + 1) * 64;
            stage_fallback(A + (size_t)bm0 * K + kn, smem + (cur ^ 1) * ASZ, 256, K, tid);
            stage_fallback(W + (size_t)bn0 * K + kn, smem + 2 * ASZ + (cur ^ 1) * BSZ, 128, K, tid);
        }
        __syncthreads();
#endif
    }

    const int crow0 = (lane >> 4) << 3;
    const int cn = lane & 15;
#pragma unroll
    for (int i = 0; i < 4; i++) {
#pragma unroll
        for (int j = 0; j < 4; j++) {
            int gn = bn0 + wn * 64 + j * 16 + cn;
            float bval = bias[gn];
#pragma unroll
            for (int e = 0; e < 8; e++) {
                int gm = bm0 + wm * 64 + i * 16 + crow0 + e;
                float v = acc[i][j][e] + bval;
                if (mode == 0) {
                    ((__bf16*)Cout)[(size_t)gm * N + gn] = (__bf16)v;
                } else {
                    int bb = gm >> 7, t = gm & 127;
                    ((float*)Cout)[((size_t)bb * Pv + 1 + t) * Dv + gn] = v;
                }
            }
        }
    }
}

// ---------------- ant attention: seq 32, dk 64; one wave per (n,h) ------------
__global__ __launch_bounds__(32) void k_attn_ant(const __bf16* __restrict__ Q,
                                                 const __bf16* __restrict__ Kb,
                                                 const __bf16* __restrict__ V,
                                                 __bf16* __restrict__ xfreq) {
    __shared__ __bf16 ks[32 * 64];
    __shared__ __bf16 vs[32 * 64];
    const int n = blockIdx.x;       // b*G+g
    const int h = blockIdx.y;
    const int lane = threadIdx.x;   // query row a
    const size_t base = (size_t)n * 32 * 1024 + h * 64;

    const uint4* krow = (const uint4*)(Kb + base + (size_t)lane * 1024);
    const uint4* vrow = (const uint4*)(V + base + (size_t)lane * 1024);
    uint4* kd = (uint4*)(ks + lane * 64);
    uint4* vd = (uint4*)(vs + lane * 64);
#pragma unroll
    for (int i = 0; i < 8; i++) { kd[i] = krow[i]; vd[i] = vrow[i]; }
    __syncthreads();

    float q[64];
    const __bf16* qrow = Q + base + (size_t)lane * 1024;
#pragma unroll
    for (int c = 0; c < 64; c++) q[c] = (float)qrow[c];

    float s[32];
    float mx = -3.4e38f;
    for (int j = 0; j < 32; j++) {
        const __bf16* kr = ks + j * 64;
        float acc = 0.f;
#pragma unroll
        for (int c = 0; c < 64; c++) acc += q[c] * (float)kr[c];
        acc *= 0.125f;
        s[j] = acc;
        mx = fmaxf(mx, acc);
    }
    float sum = 0.f;
    for (int j = 0; j < 32; j++) { s[j] = __expf(s[j] - mx); sum += s[j]; }
    float inv = 1.f / sum;

    float o[64];
#pragma unroll
    for (int c = 0; c < 64; c++) o[c] = 0.f;
    for (int j = 0; j < 32; j++) {
        float p = s[j] * inv;
        const __bf16* vr = vs + j * 64;
#pragma unroll
        for (int c = 0; c < 64; c++) o[c] += p * (float)vr[c];
    }
    size_t orow = (((size_t)(n >> 2) * 32 + lane) * 4 + (n & 3)) * 1024 + h * 64;
#pragma unroll
    for (int c = 0; c < 64; c++) xfreq[orow + c] = (__bf16)o[c];
}

// ---------------- freq attention: seq 4, dk 64; one block per n2 -------------
__global__ __launch_bounds__(64) void k_attn_freq(const __bf16* __restrict__ Q,
                                                  const __bf16* __restrict__ Kb,
                                                  const __bf16* __restrict__ V,
                                                  __bf16* __restrict__ gridflat) {
    __shared__ __bf16 ks[4 * 1024];
    __shared__ __bf16 vs[4 * 1024];
    const int n2 = blockIdx.x;       // b*A + a
    const int t = threadIdx.x;
    const size_t kb = (size_t)n2 * 4 * 1024;

    const uint4* ksrc = (const uint4*)(Kb + kb);
    const uint4* vsrc = (const uint4*)(V + kb);
    uint4* kd = (uint4*)ks;
    uint4* vd = (uint4*)vs;
    for (int i = t; i < 512; i += 64) { kd[i] = ksrc[i]; vd[i] = vsrc[i]; }
    __syncthreads();

    const int h = t >> 2, g = t & 3;
    const __bf16* qrow = Q + kb + (size_t)g * 1024 + h * 64;
    float q[64];
#pragma unroll
    for (int c = 0; c < 64; c++) q[c] = (float)qrow[c];

    float s[4];
    float mx = -3.4e38f;
    for (int j = 0; j < 4; j++) {
        const __bf16* kr = ks + j * 1024 + h * 64;
        float acc = 0.f;
#pragma unroll
        for (int c = 0; c < 64; c++) acc += q[c] * (float)kr[c];
        acc *= 0.125f;
        s[j] = acc;
        mx = fmaxf(mx, acc);
    }
    float sum = 0.f;
    for (int j = 0; j < 4; j++) { s[j] = __expf(s[j] - mx); sum += s[j]; }
    float inv = 1.f / sum;

    float o[64];
#pragma unroll
    for (int c = 0; c < 64; c++) o[c] = 0.f;
    for (int j = 0; j < 4; j++) {
        float p = s[j] * inv;
        const __bf16* vr = vs + j * 1024 + h * 64;
#pragma unroll
        for (int c = 0; c < 64; c++) o[c] += p * (float)vr[c];
    }
    size_t orow = ((size_t)(n2 >> 5) * 128 + (size_t)(n2 & 31) * 4 + g) * 1024 + h * 64;
#pragma unroll
    for (int c = 0; c < 64; c++) gridflat[orow + c] = (__bf16)o[c];
}

// ---------------- cls = x_cls + mean(grid_final) ----------------
__global__ __launch_bounds__(256) void k_mean_cls(const float* __restrict__ x,
                                                  float* __restrict__ out) {
    int idx = blockIdx.x * 256 + threadIdx.x;
    int b = idx >> 10, d = idx & 1023;
    const float* g = out + ((size_t)b * Pv + 1) * Dv + d;
    float s = 0.f;
    for (int t = 0; t < 128; t++) s += g[(size_t)t * Dv];
    out[(size_t)b * Pv * Dv + d] = x[(size_t)b * Pv * Dv + d] + s * (1.0f / 128.0f);
}

// ---------------- grid_final += cls_out ----------------
__global__ __launch_bounds__(256) void k_add_cls(float* __restrict__ out) {
    size_t idx = (size_t)blockIdx.x * 256 + threadIdx.x;
    int d = (int)(idx & 1023);
    size_t r = idx >> 10;
    int b = (int)(r >> 7), t = (int)(r & 127);
    out[((size_t)b * Pv + 1 + t) * Dv + d] += out[(size_t)b * Pv * Dv + d];
}

// =======================================================================
extern "C" void kernel_launch(void* const* d_in, const int* in_sizes, int n_in,
                              void* d_out, int out_size, void* d_ws, size_t ws_size,
                              hipStream_t stream) {
    (void)in_sizes; (void)n_in; (void)out_size; (void)ws_size;

    const float* x = (const float*)d_in[0];
    const float* w[7]  = {(const float*)d_in[1], (const float*)d_in[3],
                          (const float*)d_in[5], (const float*)d_in[7],
                          (const float*)d_in[9], (const float*)d_in[11],
                          (const float*)d_in[13]};
    const float* bs[7] = {(const float*)d_in[2], (const float*)d_in[4],
                          (const float*)d_in[6], (const float*)d_in[8],
                          (const float*)d_in[10], (const float*)d_in[12],
                          (const float*)d_in[14]};

    char* ws = (char*)d_ws;
    const size_t WBYTES = 2ull * 1024 * 1024;
    const size_t BUF    = (size_t)MROWS * 1024 * 2;
    __bf16* wb[7];
    for (int i = 0; i < 7; i++) wb[i] = (__bf16*)(ws + (size_t)i * WBYTES);
    __bf16* XA = (__bf16*)(ws + (16ull << 20));
    __bf16* Qb = (__bf16*)((char*)XA + 1 * BUF);
    __bf16* Kb = (__bf16*)((char*)XA + 2 * BUF);
    __bf16* Vb = (__bf16*)((char*)XA + 3 * BUF);
    __bf16* XF = (__bf16*)((char*)XA + 4 * BUF);

    for (int i = 0; i < 7; i++)
        k_cvt<<<4096, 256, 0, stream>>>(w[i], wb[i], 1024 * 1024);

    k_permute_ant<<<262144, 256, 0, stream>>>(x, XA);

    dim3 gg(MROWS / 256, 1024 / 128);

    k_gemm_bf16<<<gg, 256, 0, stream>>>(XA, wb[0], bs[0], Qb, MROWS, 1024, 1024, 0);
    k_gemm_bf16<<<gg, 256, 0, stream>>>(XA, wb[1], bs[1], Kb, MROWS, 1024, 1024, 0);
    k_gemm_bf16<<<gg, 256, 0, stream>>>(XA, wb[2], bs[2], Vb, MROWS, 1024, 1024, 0);

    dim3 ga(2048, 16);
    k_attn_ant<<<ga, 32, 0, stream>>>(Qb, Kb, Vb, XF);

    k_gemm_bf16<<<gg, 256, 0, stream>>>(XF, wb[3], bs[3], Qb, MROWS, 1024, 1024, 0);
    k_gemm_bf16<<<gg, 256, 0, stream>>>(XF, wb[4], bs[4], Kb, MROWS, 1024, 1024, 0);
    k_gemm_bf16<<<gg, 256, 0, stream>>>(XF, wb[5], bs[5], Vb, MROWS, 1024, 1024, 0);

    k_attn_freq<<<16384, 64, 0, stream>>>(Qb, Kb, Vb, XA);

    k_gemm_bf16<<<gg, 256, 0, stream>>>(XA, wb[6], bs[6], d_out, MROWS, 1024, 1024, 1);

    k_mean_cls<<<2048, 256, 0, stream>>>(x, (float*)d_out);
    k_add_cls<<<262144, 256, 0, stream>>>((float*)d_out);
}
